// DeformableAttention3D_18305150615727
// MI455X (gfx1250) — matrix operand order repaired and checked by reading, not compiled, not run on hardware
//
#include <hip/hip_runtime.h>
#include <hip/hip_bf16.h>
#include <math.h>

typedef __attribute__((ext_vector_type(16))) _Float16 v16h;
typedef __attribute__((ext_vector_type(8)))  _Float16 v8h;
typedef __attribute__((ext_vector_type(8)))  float    v8f;

#define EMBED    256
#define HEADS    8
#define HEAD_DIM 32
#define LEVELS   4
#define POINTS   4
#define BATCH    2
#define NQ       10000
#define MTOT     21760

// ---------------------------------------------------------------------------
// Prep: transpose + convert weights  W[K=256][NCOLS] f32  ->  Wt[NCOLS][256] f16
// Tiny (<=256 KB), runs once; makes the GEMM B-fragment a contiguous 32B read.
// ---------------------------------------------------------------------------
__global__ __launch_bounds__(256)
void transpose_cvt_f16(const float* __restrict__ W, _Float16* __restrict__ Wt,
                       int ncols)
{
    const int id = blockIdx.x * 256 + threadIdx.x;   // id < ncols*256
    const int n  = id >> 8;          // output row   (0..ncols-1)
    const int k  = id & 255;         // output col   (0..255)
    Wt[id] = (_Float16)W[(size_t)k * ncols + n];
}

// ---------------------------------------------------------------------------
// WMMA GEMM:  Y[R, NCOLS] = X[R, 256] @ W[256, NCOLS] + bias
//   - block = NCOLS/16 waves, one 16x16 output tile per wave
//   - A tile (16 rows x 256 K) cooperatively converted f32->f16 into LDS once,
//     fragments pulled with ds_load_b128
//   - B fragments read from pre-transposed f16 Wt as contiguous b128 loads
// ---------------------------------------------------------------------------
template<int NCOLS>
__global__ __launch_bounds__((NCOLS / 16) * 32)
void gemm_bias_wmma(const float* __restrict__ X,
                    const _Float16* __restrict__ Wt,   // [NCOLS][256] f16
                    const float* __restrict__ bias,
                    float* __restrict__ Y)
{
    constexpr int NTHREADS = (NCOLS / 16) * 32;
    __shared__ _Float16 xs[16 * EMBED];               // 8 KB

    const int tid   = threadIdx.x;
    const int lane  = tid & 31;
    const int wave  = tid >> 5;
    const int rb    = blockIdx.x * 16;                // row tile base
    const int cb    = wave * 16;                      // col tile base
    const int lhalf = lane & 15;
    const bool hi   = (lane >= 16);

    // ---- cooperative f32->f16 conversion of the 16x256 A tile into LDS ----
    // 4096 halfs = 512 chunks of 8; each thread converts chunks strided by NTHREADS
#pragma unroll
    for (int c = tid; c < 512; c += NTHREADS) {
        const int flat = c * 8;                        // row-major over [16][256]
        const int row  = flat >> 8;
        const int col  = flat & 255;
        const float4 f0 = *(const float4*)(X + (size_t)(rb + row) * EMBED + col);
        const float4 f1 = *(const float4*)(X + (size_t)(rb + row) * EMBED + col + 4);
        v8h h;
        h[0]=(_Float16)f0.x; h[1]=(_Float16)f0.y; h[2]=(_Float16)f0.z; h[3]=(_Float16)f0.w;
        h[4]=(_Float16)f1.x; h[5]=(_Float16)f1.y; h[6]=(_Float16)f1.z; h[7]=(_Float16)f1.w;
        *(v8h*)(xs + flat) = h;                        // ds_store_b128
    }
    __syncthreads();

    const int koffA = hi ? 16 : 0;   // A frag: lanes<16 K 0..15; lanes>=16 K 16..31
    const int koffB = hi ? 16 : 0;   // B frag: lanes<16 K 0..15; lanes>=16 K 16..31
    const int ncol  = cb + lhalf;
    const _Float16* wrow = Wt + (size_t)ncol * EMBED;

    v8f acc = {};
#pragma unroll
    for (int kb = 0; kb < EMBED; kb += 32) {
        // A fragment: two ds_load_b128 from LDS
        const v8h alo = *(const v8h*)(xs + lhalf * EMBED + kb + koffA);
        const v8h ahi = *(const v8h*)(xs + lhalf * EMBED + kb + 8 + koffA);
        v16h a;
#pragma unroll
        for (int i = 0; i < 8; ++i) { a[i] = alo[i]; a[8 + i] = ahi[i]; }

        // B fragment: 16 contiguous f16 (32B) from transposed weights
        const v16h b = *(const v16h*)(wrow + kb + koffB);

        acc = __builtin_amdgcn_wmma_f32_16x16x32_f16(
                  false, a, false, b, (short)0, acc, false, false);
    }

    // C/D layout: VGPR v -> M=v (lanes 0-15) / M=8+v (lanes 16-31); N = cb+lhalf
    const int mbase = hi ? 8 : 0;
    const float bv  = bias[ncol];
#pragma unroll
    for (int v = 0; v < 8; ++v) {
        Y[(size_t)(rb + mbase + v) * NCOLS + ncol] = acc[v] + bv;
    }
}

// ---------------------------------------------------------------------------
// Deformable sampling: one wave per (b, n, h); lane = head-dim channel.
// Corner gathers are 32 consecutive floats -> coalesced 128B loads, served
// from L2 (v_proj = 44.6 MB < 192 MB L2).
// ---------------------------------------------------------------------------
__global__ __launch_bounds__(256)
void deform_sample(const float* __restrict__ vproj,   // (B, M, 256)
                   const float* __restrict__ off,     // (B, N, 256)
                   const float* __restrict__ logit,   // (B, N, 128)
                   const float* __restrict__ ref,     // (B, N, 1, 2)
                   float* __restrict__ out)           // (B, N, 256)
{
    const int lane = threadIdx.x & 31;
    const int wid  = blockIdx.x * 8 + (threadIdx.x >> 5);   // [0, B*N*H)
    const int h    = wid % HEADS;
    const int bn   = wid / HEADS;                           // b*N + n
    const int b    = bn / NQ;

    const float rx = ref[(size_t)bn * 2 + 0];
    const float ry = ref[(size_t)bn * 2 + 1];

    // softmax over the 16 (level,point) logits for this head (uniform per wave)
    const float* lg = logit + (size_t)bn * (HEADS * 16) + h * 16;
    float lv[16];
    float mx = -1e30f;
#pragma unroll
    for (int i = 0; i < 16; ++i) { lv[i] = lg[i]; mx = fmaxf(mx, lv[i]); }
    float sum = 0.f;
#pragma unroll
    for (int i = 0; i < 16; ++i) { lv[i] = __expf(lv[i] - mx); sum += lv[i]; }
    const float inv = 1.0f / sum;

    const float* op    = off + (size_t)bn * EMBED + h * (LEVELS * POINTS * 2);
    const float* vbase = vproj + (size_t)b * MTOT * EMBED + h * HEAD_DIM + lane;

    const int lvlH[4]   = {128, 64, 32, 16};
    const int lvlW[4]   = {128, 64, 32, 16};
    const int starts[4] = {0, 16384, 20480, 21504};

    float acc = 0.f;
#pragma unroll
    for (int l = 0; l < LEVELS; ++l) {
        const int   Hl = lvlH[l], Wl = lvlW[l];
        const float fW = (float)Wl, fH = (float)Hl;
        const float* vl = vbase + (size_t)starts[l] * EMBED;
#pragma unroll
        for (int p = 0; p < POINTS; ++p) {
            const float ox = op[l * (POINTS * 2) + p * 2 + 0];
            const float oy = op[l * (POINTS * 2) + p * 2 + 1];
            const float x  = (rx + ox / fW) * fW - 0.5f;
            const float y  = (ry + oy / fH) * fH - 0.5f;
            const float x0f = floorf(x), y0f = floorf(y);
            const float lx = x - x0f, ly = y - y0f;
            const int   x0 = (int)x0f, y0 = (int)y0f;
            const float aw = lv[l * POINTS + p] * inv;

            float s = 0.f;
#pragma unroll
            for (int c = 0; c < 4; ++c) {
                const int   dx = c & 1, dy = c >> 1;
                const float w  = (dx ? lx : 1.f - lx) * (dy ? ly : 1.f - ly);
                const int   xi = x0 + dx, yi = y0 + dy;
                if (xi >= 0 && xi < Wl && yi >= 0 && yi < Hl) {
                    s += w * vl[(size_t)(yi * Wl + xi) * EMBED];
                }
            }
            acc += aw * s;
        }
    }

    out[(size_t)bn * EMBED + h * HEAD_DIM + lane] = acc;
}

// ---------------------------------------------------------------------------
extern "C" void kernel_launch(void* const* d_in, const int* in_sizes, int n_in,
                              void* d_out, int out_size, void* d_ws, size_t ws_size,
                              hipStream_t stream)
{
    const float* query  = (const float*)d_in[0];
    // d_in[1] = key (unused by reference)
    const float* value  = (const float*)d_in[2];
    const float* ref    = (const float*)d_in[3];
    // d_in[4] spatial_shapes, d_in[5] level_start_idx: compile-time constants
    const float* W_off  = (const float*)d_in[6];
    const float* b_off  = (const float*)d_in[7];
    const float* W_attn = (const float*)d_in[8];
    const float* b_attn = (const float*)d_in[9];
    const float* W_v    = (const float*)d_in[10];
    const float* b_v    = (const float*)d_in[11];
    float* out = (float*)d_out;

    const size_t VPROJ_BYTES = (size_t)BATCH * MTOT * EMBED * sizeof(float); // 44.56 MB
    const size_t OFF_BYTES   = (size_t)BATCH * NQ * EMBED * sizeof(float);   // 20.48 MB
    const size_t LOGIT_BYTES = (size_t)BATCH * NQ * (HEADS * 16) * sizeof(float);

    char*      ws      = (char*)d_ws;
    float*     vproj   = (float*)ws;
    float*     offb    = (float*)(ws + VPROJ_BYTES);
    float*     logit   = (float*)(ws + VPROJ_BYTES + OFF_BYTES);
    _Float16*  Wt_v    = (_Float16*)(ws + VPROJ_BYTES + OFF_BYTES + LOGIT_BYTES);
    _Float16*  Wt_off  = Wt_v   + 256 * 256;
    _Float16*  Wt_attn = Wt_off + 256 * 256;

    // 0) one-time weight transpose + f32->f16 conversion
    transpose_cvt_f16<<<(256 * 256) / 256, 256, 0, stream>>>(W_v,    Wt_v,    256);
    transpose_cvt_f16<<<(256 * 256) / 256, 256, 0, stream>>>(W_off,  Wt_off,  256);
    transpose_cvt_f16<<<(128 * 256) / 256, 256, 0, stream>>>(W_attn, Wt_attn, 128);

    // 1) value projection: (B*M, 256) @ (256,256)
    gemm_bias_wmma<256><<<(BATCH * MTOT) / 16, 512, 0, stream>>>(value, Wt_v, b_v, vproj);
    // 2) sampling offsets: (B*N, 256) @ (256,256)
    gemm_bias_wmma<256><<<(BATCH * NQ) / 16, 512, 0, stream>>>(query, Wt_off, b_off, offb);
    // 3) attention logits: (B*N, 256) @ (256,128)
    gemm_bias_wmma<128><<<(BATCH * NQ) / 16, 256, 0, stream>>>(query, Wt_attn, b_attn, logit);
    // 4) softmax + bilinear gather + weighted reduce
    deform_sample<<<(BATCH * NQ * HEADS) / 8, 256, 0, stream>>>(vproj, offb, logit, ref, out);
}
